// BiLSTM_Model_59588376264813
// MI455X (gfx1250) — matrix operand order repaired and checked by reading, compiled, not run on hardware
//
#include <hip/hip_runtime.h>
#include <hip/hip_bf16.h>

// ---------------------------------------------------------------------------
// BiLSTM for MI455X (gfx1250).  B=128 T=256 V=50000 E=300 H=256.
//   K0: bf16 conversion of emb / W_ih (K padded 300->320) / W_hh, bias fold.
//   K1: xg GEMM (32768 x 1024 x 320) x 2 dirs, v_wmma_f32_16x16x32_bf16,
//       embedding gather fused into A-fragment loads, double-buffered k-loop.
//   K2: persistent recurrence, one WG per (dir, 16 batch rows) -- batch rows
//       are independent, so no cross-WG sync.  h stays in LDS, c in registers,
//       W_hh streams from L2, xg tile staged to LDS via async load-to-LDS.
// ---------------------------------------------------------------------------

#define B_  128
#define T_  256
#define V_  50000
#define E_  300
#define H_  256
#define G4  (4 * H_)      // 1024
#define KP  320           // E padded to multiple of 32

typedef __attribute__((ext_vector_type(16))) __bf16         bf16x16;
typedef __attribute__((ext_vector_type(8)))  float          f32x8;
typedef __attribute__((ext_vector_type(8)))  unsigned short u16x8;
typedef __attribute__((ext_vector_type(16))) unsigned short u16x16;
typedef __attribute__((ext_vector_type(4)))  int            i32x4;

#if defined(__AMDGCN__) && __has_builtin(__builtin_amdgcn_global_load_async_to_lds_b128)
#define HAVE_ASYNC_LDS 1
#endif

#if defined(HAVE_ASYNC_LDS)
typedef __attribute__((address_space(1))) i32x4 i32x4_g;   // global
typedef __attribute__((address_space(3))) i32x4 i32x4_l;   // LDS
__device__ __forceinline__ void async_b128(const void* g, void* l) {
  __builtin_amdgcn_global_load_async_to_lds_b128(
      (i32x4_g*)g, (i32x4_l*)l, 0, 0);
}
__device__ __forceinline__ void wait_async0() {
#if __has_builtin(__builtin_amdgcn_s_wait_asynccnt)
  __builtin_amdgcn_s_wait_asynccnt(0);
#else
  asm volatile("s_wait_asynccnt 0x0" ::: "memory");
#endif
}
#endif

__device__ __forceinline__ unsigned short f2bf(float f) {
  unsigned u = __builtin_bit_cast(unsigned, f);
  unsigned r = 0x7FFFu + ((u >> 16) & 1u);       // round-to-nearest-even
  return (unsigned short)((u + r) >> 16);
}

// A fragment (16x32 bf16, MxK).  rowp = this lane's row pointer.
// lanes 0-15: elems 0..7 -> K 0..7,  elems 8..15 -> K 16..23
// lanes 16-31: elems 0..7 -> K 8..15, elems 8..15 -> K 24..31
__device__ __forceinline__ bf16x16 frag_a_row(const unsigned short* rowp,
                                              int kofs, int kh) {
  const unsigned short* p = rowp + kofs + kh * 8;
  u16x8 lo = *(const u16x8*)(p);        // 16B load
  u16x8 hi = *(const u16x8*)(p + 16);   // 16B load
  u16x16 v;
#pragma unroll
  for (int i = 0; i < 8; ++i) { v[i] = lo[i]; v[i + 8] = hi[i]; }
  return __builtin_bit_cast(bf16x16, v);
}

// B fragment (32x16 bf16, KxN) taken from row-major W[N][K]: rowp = row n.
// lanes 0-15: K 0..7,16..23 ; lanes 16-31: K 8..15,24..31
__device__ __forceinline__ bf16x16 frag_b_row(const unsigned short* rowp,
                                              int kofs, int kh) {
  const unsigned short* p = rowp + kofs + kh * 8;
  u16x8 lo = *(const u16x8*)(p);
  u16x8 hi = *(const u16x8*)(p + 16);
  u16x16 v;
#pragma unroll
  for (int i = 0; i < 8; ++i) { v[i] = lo[i]; v[i + 8] = hi[i]; }
  return __builtin_bit_cast(bf16x16, v);
}

// ---------------------------------------------------------------------------
// K0: prep -- bf16 conversions + bias folding
// ---------------------------------------------------------------------------
__global__ void __launch_bounds__(256)
prep_kernel(const float* __restrict__ emb,
            const float* __restrict__ Wih_fw, const float* __restrict__ Whh_fw,
            const float* __restrict__ bih_fw, const float* __restrict__ bhh_fw,
            const float* __restrict__ Wih_bw, const float* __restrict__ Whh_bw,
            const float* __restrict__ bih_bw, const float* __restrict__ bhh_bw,
            unsigned short* __restrict__ embT,
            unsigned short* __restrict__ W_ihT,
            unsigned short* __restrict__ W_hhB,
            float* __restrict__ b_sum) {
  size_t idx    = (size_t)blockIdx.x * blockDim.x + threadIdx.x;
  size_t stride = (size_t)gridDim.x * blockDim.x;

  const size_t nEmb = (size_t)V_ * KP;
  for (size_t i = idx; i < nEmb; i += stride) {
    int    k = (int)(i % KP);
    size_t v = i / KP;
    embT[i] = (k < E_) ? f2bf(emb[v * E_ + k]) : (unsigned short)0;
  }
  const size_t nWih = (size_t)G4 * KP;
  for (size_t i = idx; i < nWih; i += stride) {
    int    k = (int)(i % KP);
    size_t n = i / KP;
    W_ihT[i]        = (k < E_) ? f2bf(Wih_fw[n * E_ + k]) : (unsigned short)0;
    W_ihT[nWih + i] = (k < E_) ? f2bf(Wih_bw[n * E_ + k]) : (unsigned short)0;
  }
  const size_t nWhh = (size_t)G4 * H_;
  for (size_t i = idx; i < nWhh; i += stride) {
    W_hhB[i]        = f2bf(Whh_fw[i]);
    W_hhB[nWhh + i] = f2bf(Whh_bw[i]);
  }
  for (size_t i = idx; i < G4; i += stride) {
    b_sum[i]      = bih_fw[i] + bhh_fw[i];
    b_sum[G4 + i] = bih_bw[i] + bhh_bw[i];
  }
}

// ---------------------------------------------------------------------------
// K1: xg = gather(emb,x) @ W_ih^T + (b_ih + b_hh)     [2][T*B][1024]
// One 16x64 strip per wave; K=320 -> 10 WMMA k-steps, double-buffered loads.
// ---------------------------------------------------------------------------
__global__ void __launch_bounds__(256)
xg_gemm_kernel(const int* __restrict__ x,
               const unsigned short* __restrict__ embT,
               const unsigned short* __restrict__ W_ihT,
               const float* __restrict__ b_sum,
               float* __restrict__ xg) {
  const int wavesPerBlock = blockDim.x >> 5;
  const int gw   = blockIdx.x * wavesPerBlock + ((int)threadIdx.x >> 5);
  const int lane = (int)threadIdx.x & 31;

  const int nStrips = G4 / 64;          // 16
  const int mStrips = (T_ * B_) / 16;   // 2048
  int dir = gw / (mStrips * nStrips);
  if (dir >= 2) return;
  int rem = gw % (mStrips * nStrips);
  int ms  = rem / nStrips;
  int ns  = rem % nStrips;

  const int col = lane & 15, kh = lane >> 4;

  // A row for this lane: m = t*B + b
  int m     = ms * 16 + col;
  int b     = m % B_;
  int t     = m / B_;
  int token = x[b * T_ + t];
  const unsigned short* arow  = embT + (size_t)token * KP;
  const unsigned short* wbase = W_ihT + (size_t)dir * G4 * KP + (size_t)(ns * 64) * KP;
  const unsigned short* brow[4];
#pragma unroll
  for (int nt = 0; nt < 4; ++nt) brow[nt] = wbase + (size_t)(nt * 16 + col) * KP;

  f32x8 acc[4] = {};
  bf16x16 aCur = frag_a_row(arow, 0, kh);
  bf16x16 bCur[4];
#pragma unroll
  for (int nt = 0; nt < 4; ++nt) bCur[nt] = frag_b_row(brow[nt], 0, kh);

#pragma unroll
  for (int kk = 0; kk < KP / 32; ++kk) {
    bf16x16 aNxt = aCur;
    bf16x16 bNxt[4];
    if (kk + 1 < KP / 32) {                 // prefetch next k-step
      aNxt = frag_a_row(arow, (kk + 1) * 32, kh);
#pragma unroll
      for (int nt = 0; nt < 4; ++nt)
        bNxt[nt] = frag_b_row(brow[nt], (kk + 1) * 32, kh);
    } else {
#pragma unroll
      for (int nt = 0; nt < 4; ++nt) bNxt[nt] = bCur[nt];
    }
#pragma unroll
    for (int nt = 0; nt < 4; ++nt)
      acc[nt] = __builtin_amdgcn_wmma_f32_16x16x32_bf16(
          false, aCur, false, bCur[nt], (short)0, acc[nt], false, false);
    aCur = aNxt;
#pragma unroll
    for (int nt = 0; nt < 4; ++nt) bCur[nt] = bNxt[nt];
  }

  float*       outb = xg + (size_t)dir * ((size_t)T_ * B_ * G4)
                         + (size_t)(ms * 16) * G4 + ns * 64;
  const float* bs   = b_sum + dir * G4 + ns * 64;
#pragma unroll
  for (int nt = 0; nt < 4; ++nt) {
    float bias = bs[nt * 16 + col];
#pragma unroll
    for (int i = 0; i < 8; ++i) {
      int row = kh * 8 + i;                       // C/D layout: M = 8*(l>>4)+i
      outb[(size_t)row * G4 + nt * 16 + col] = acc[nt][i] + bias;
    }
  }
}

// ---------------------------------------------------------------------------
// K2: persistent LSTM scan.  grid = 2 dirs x 8 row-groups, 256 thr (8 waves).
// LDS: gates 16x1024 f32 (64KB) | h 16x256 bf16 (8KB) | len (64B) | xg 64KB
// c state lives in registers: thread tid owns column j=tid for rows 0..15.
// ---------------------------------------------------------------------------
__global__ void __launch_bounds__(256)
lstm_scan_kernel(const int* __restrict__ lengths,
                 const unsigned short* __restrict__ W_hhB,
                 const float* __restrict__ xg,
                 float* __restrict__ out) {
  extern __shared__ char smem[];
  float*          gatesS = (float*)smem;                          // 64KB
  unsigned short* hS     = (unsigned short*)(smem + 65536);       // 8KB
  int*            lenS   = (int*)(smem + 65536 + 8192);           // 64B
  float*          xgS    = (float*)(smem + 65536 + 8192 + 1024);  // 64KB

  const int dir     = blockIdx.x >> 3;
  const int rowBase = (blockIdx.x & 7) * 16;     // batch-row slice
  const int tid  = (int)threadIdx.x;
  const int lane = tid & 31, wid = tid >> 5;
  const int col  = lane & 15, kh = lane >> 4;
  const int nBase = wid * 128;                   // 8 n-tiles per wave

  for (int i = tid; i < 16 * H_; i += blockDim.x) hS[i] = 0;
  if (tid < 16) lenS[tid] = lengths[rowBase + tid];
  float cReg[16];
#pragma unroll
  for (int r = 0; r < 16; ++r) cReg[r] = 0.0f;
  __syncthreads();

  const unsigned short* whh = W_hhB + (size_t)dir * G4 * H_;
  const float*          xgd = xg + (size_t)dir * ((size_t)T_ * B_ * G4);
  const unsigned short* browS[8];
#pragma unroll
  for (int nt = 0; nt < 8; ++nt)
    browS[nt] = whh + (size_t)(nBase + nt * 16 + col) * H_;

  for (int step = 0; step < T_; ++step) {
    const int t = dir ? (T_ - 1 - step) : step;
    const float* xgt = xgd + ((size_t)t * B_ + rowBase) * G4;

#if defined(HAVE_ASYNC_LDS)
    // Stage this step's 64KB xg tile into LDS asynchronously (ASYNCcnt);
    // the copy overlaps the gate GEMM below.
    {
      const char* src = (const char*)xgt;
      char*       dst = (char*)xgS;
#pragma unroll
      for (int j = 0; j < 16; ++j) {
        int ofs = (tid + j * 256) * 16;
        async_b128(src + ofs, dst + ofs);
      }
    }
#endif
    // Prefetch next step's xg tile toward L2 (global_prefetch_b8).
    {
      int tn = dir ? (t - 1) : (t + 1);
      if (tn >= 0 && tn < T_) {
        const char* p = (const char*)(xgd + ((size_t)tn * B_ + rowBase) * G4);
        __builtin_prefetch(p + tid * 256, 0, 1);
      }
    }

    // gates(16x1024) = h(16x256) @ W_hh^T.  A-frags hoisted (step-invariant
    // in k), B-frags one-deep software pipeline from L2-resident W_hh.
    bf16x16 afr[8];
#pragma unroll
    for (int kk = 0; kk < 8; ++kk)
      afr[kk] = frag_a_row(hS + (size_t)col * H_, kk * 32, kh);

    f32x8 acc[8] = {};
    bf16x16 bCur = frag_b_row(browS[0], 0, kh);
#pragma unroll
    for (int it = 0; it < 64; ++it) {
      int kk = it >> 3, nt = it & 7;
      bf16x16 bNxt = bCur;
      if (it + 1 < 64) {
        int kk2 = (it + 1) >> 3, nt2 = (it + 1) & 7;
        bNxt = frag_b_row(browS[nt2], kk2 * 32, kh);
      }
      acc[nt] = __builtin_amdgcn_wmma_f32_16x16x32_bf16(
          false, afr[kk], false, bCur, (short)0, acc[nt], false, false);
      bCur = bNxt;
    }
#pragma unroll
    for (int nt = 0; nt < 8; ++nt)
#pragma unroll
      for (int i = 0; i < 8; ++i)
        gatesS[(kh * 8 + i) * G4 + nBase + nt * 16 + col] = acc[nt][i];

#if defined(HAVE_ASYNC_LDS)
    wait_async0();
#endif
    __syncthreads();

    // elementwise LSTM cell: thread tid owns column j=tid, rows 0..15
    const int j = tid;
#pragma unroll
    for (int r = 0; r < 16; ++r) {
#if defined(HAVE_ASYNC_LDS)
      const float* xr = xgS + (size_t)r * G4;
#else
      const float* xr = xgt + (size_t)r * G4;
#endif
      int gb = r * G4;
      float gi = gatesS[gb + j]          + xr[j];
      float gf = gatesS[gb + H_ + j]     + xr[H_ + j];
      float gg = gatesS[gb + 2 * H_ + j] + xr[2 * H_ + j];
      float go = gatesS[gb + 3 * H_ + j] + xr[3 * H_ + j];
      float si = 1.0f / (1.0f + __expf(-gi));
      float sf = 1.0f / (1.0f + __expf(-gf));
      float so = 1.0f / (1.0f + __expf(-go));
      float tg = tanhf(gg);
      float c  = sf * cReg[r] + si * tg;
      float h  = so * tanhf(c);
      float mk = (t < lenS[r]) ? 1.0f : 0.0f;
      c *= mk;  h *= mk;
      cReg[r] = c;
      hS[r * H_ + j] = f2bf(h);
      out[(((size_t)(rowBase + r)) * T_ + t) * (2 * H_) + dir * H_ + j] = h;
    }
    __syncthreads();
  }
}

// ---------------------------------------------------------------------------
extern "C" void kernel_launch(void* const* d_in, const int* in_sizes, int n_in,
                              void* d_out, int out_size, void* d_ws, size_t ws_size,
                              hipStream_t stream) {
  const int*   x       = (const int*)d_in[0];
  const int*   lengths = (const int*)d_in[1];
  const float* emb     = (const float*)d_in[2];
  const float* Wih_fw  = (const float*)d_in[3];
  const float* Whh_fw  = (const float*)d_in[4];
  const float* bih_fw  = (const float*)d_in[5];
  const float* bhh_fw  = (const float*)d_in[6];
  const float* Wih_bw  = (const float*)d_in[7];
  const float* Whh_bw  = (const float*)d_in[8];
  const float* bih_bw  = (const float*)d_in[9];
  const float* bhh_bw  = (const float*)d_in[10];
  float*       out     = (float*)d_out;

  size_t off = 0;
  auto carve = [&](size_t bytes) {
    void* p = (char*)d_ws + off;
    off += (bytes + 255) & ~(size_t)255;
    return p;
  };
  unsigned short* embT  = (unsigned short*)carve((size_t)V_ * KP * 2);
  unsigned short* W_ihT = (unsigned short*)carve((size_t)2 * G4 * KP * 2);
  unsigned short* W_hhB = (unsigned short*)carve((size_t)2 * G4 * H_ * 2);
  float*          b_sum = (float*)carve((size_t)2 * G4 * 4);
  float*          xgbuf = (float*)carve((size_t)2 * T_ * B_ * G4 * 4);

  prep_kernel<<<2048, 256, 0, stream>>>(emb, Wih_fw, Whh_fw, bih_fw, bhh_fw,
                                        Wih_bw, Whh_bw, bih_bw, bhh_bw,
                                        embT, W_ihT, W_hhB, b_sum);

  // 2 dirs * 2048 m-strips * 16 n-strips = 65536 wave-strips / 8 waves per blk
  xg_gemm_kernel<<<8192, 256, 0, stream>>>(x, embT, W_ihT, b_sum, xgbuf);

  // LDS: 64KB gates + 8KB h + 1KB len + 64KB xg stage = 137KB (+pad)
  lstm_scan_kernel<<<16, 256, 140 * 1024, stream>>>(lengths, W_hhB, xgbuf, out);
}
